// RegionProposalNetwork_27453430956787
// MI455X (gfx1250) — matrix operand order repaired and checked by reading, not compiled, not run on hardware
//
#include <hip/hip_runtime.h>
#include <hip/hip_bf16.h>
#include <math.h>

typedef unsigned short u16;
typedef unsigned int   u32;

typedef __attribute__((ext_vector_type(16))) __bf16 v16bf;
typedef __attribute__((ext_vector_type(8)))  float  v8f;

struct __align__(16) B16x8  { u32 d[4]; };  // 8 bf16
struct __align__(32) B16x16 { u32 d[8]; };  // 16 bf16 (one WMMA operand)

#define HH    100
#define WW    152
#define NPIX  15200            // 100*152
#define CIN   1024
#define COUT  512
#define PH    102              // padded height
#define PW    154              // padded width
#define NA    15               // anchors per pixel
#define NANCH 228000           // NPIX*NA
#define SORTN 262144           // 2^18 >= NANCH
#define NPRE  6000
#define NPOST 300
#define N2    96               // 60 loc + 30 score + 6 pad

// d_out flat layout (floats), reference tuple order
#define OFF_LOCS   0
#define OFF_SCORES 912000
#define OFF_ROIS   1368000
#define OFF_RIDX   1369200
#define OFF_ANCHOR 1369500

__device__ __forceinline__ u16 f2bf(float f) {
    u32 u = __float_as_uint(f);
    u += 0x7FFFu + ((u >> 16) & 1u);          // round-to-nearest-even
    return (u16)(u >> 16);
}

// ---------------- CDNA5 async memory->LDS helpers ----------------
// global_load_async_to_lds_b128: per-lane 16B copy, tracked by ASYNCcnt.
// ISA: INST_OFFSET is added to BOTH the LDS dest and the global source, so a
// 64B copy is 4 asyncs sharing one address pair with offset:0/16/32/48.
__device__ __forceinline__ void async_copy64(u32 lds_off, const u16* g) {
    unsigned long long ga = (unsigned long long)(size_t)g;
    asm volatile("global_load_async_to_lds_b128 %0, %1, off\n\t"
                 "global_load_async_to_lds_b128 %0, %1, off offset:16\n\t"
                 "global_load_async_to_lds_b128 %0, %1, off offset:32\n\t"
                 "global_load_async_to_lds_b128 %0, %1, off offset:48"
                 :: "v"(lds_off), "v"(ga)
                 : "memory");
}
__device__ __forceinline__ void wait_async0() {
    asm volatile("s_wait_asynccnt 0x0" ::: "memory");
}
__device__ __forceinline__ u32 lds_addr(const void* p) {
    return (u32)(size_t)p;    // flat LDS pointer: low 32 bits are LDS offset
}

// ---------------- repack kernels ----------------

// x: NCHW fp32 (1024,100,152) -> zero-padded NHWC bf16 (102,154,1024)
__global__ void k_pad_x(const float* __restrict__ x, u16* __restrict__ xpad) {
    int t = blockIdx.x * 256 + threadIdx.x;
    if (t >= PH * PW * CIN) return;
    int c = t & (CIN - 1);
    int pp = t >> 10;
    int y = pp / PW, xx = pp % PW;
    float v = 0.0f;
    if (y >= 1 && y <= HH && xx >= 1 && xx <= WW)
        v = x[c * NPIX + (y - 1) * WW + (xx - 1)];
    xpad[t] = f2bf(v);
}

// conv1_w OIHW (512,1024,3,3) fp32 -> w1t[tap][n][c] bf16
__global__ void k_pack_w1(const float* __restrict__ w, u16* __restrict__ w1t) {
    int t = blockIdx.x * 256 + threadIdx.x;
    if (t >= 9 * COUT * CIN) return;
    int c = t & (CIN - 1);
    int rem = t >> 10;
    int n = rem & (COUT - 1);
    int tap = rem >> 9;
    w1t[t] = f2bf(w[(n * CIN + c) * 9 + tap]);
}

// concat loc_w(60,512) and score_w(30,512) -> w2t[n][c] bf16 (96x512)
__global__ void k_pack_w2(const float* __restrict__ loc_w, const float* __restrict__ score_w,
                          u16* __restrict__ w2t) {
    int t = blockIdx.x * 256 + threadIdx.x;
    if (t >= N2 * COUT) return;
    int c = t & (COUT - 1);
    int n = t >> 9;
    float v = 0.0f;
    if (n < 60)      v = loc_w[n * COUT + c];
    else if (n < 90) v = score_w[(n - 60) * COUT + c];
    w2t[t] = f2bf(v);
}

// ---------------- WMMA fragment helpers ----------------
// A 16x32 bf16:              lane&15 = M, K = (lane>>4)*16 .. +15 contiguous
// B 32x16 bf16:              lane&15 = N, K = (lane>>4)*16 .. +15 contiguous
// C 16x16 f32:               lane&15 = N, VGPR r -> M = r + (lane>>4)*8

__device__ __forceinline__ v16bf load_a_frag(const u16* row_ptr, int lane) {
    int ks = (lane >> 4) * 16;
    B16x16 t = *reinterpret_cast<const B16x16*>(row_ptr + ks);
    return __builtin_bit_cast(v16bf, t);
}
__device__ __forceinline__ v16bf load_b_frag(const u16* col_ptr, int lane) {
    int kb = (lane >> 4) * 16;
    B16x16 t = *reinterpret_cast<const B16x16*>(col_ptr + kb);
    return __builtin_bit_cast(v16bf, t);
}

// ---------------- GEMM1: 3x3 conv via implicit GEMM ----------------
// M=15200 (pixels), N=512, K = 9 taps x 1024 ch. Tile 128x128, 8 waves (4x2).
// Double-buffered LDS (K=64 per step), async copies overlap WMMA compute.
__global__ __launch_bounds__(256)
void k_conv3x3_wmma(const u16* __restrict__ xpad, const u16* __restrict__ w1t,
                    const float* __restrict__ bias, u16* __restrict__ hout) {
    __shared__ u16 As[2][128 * 64];   // [pixel][c]
    __shared__ u16 Bs[2][128 * 64];   // [n][c]
    const int tid  = threadIdx.x;
    const int lane = tid & 31;
    const int wv   = tid >> 5;
    const int wm   = wv >> 1;                 // 0..3
    const int wn   = wv & 1;                  // 0..1
    const int pbase = blockIdx.x * 128;
    const int nbase = blockIdx.y * 128;

    const int r    = tid >> 1;                // staged row 0..127
    const int half = tid & 1;                 // which 32-ch half
    const int p    = pbase + r;
    const bool pv  = (p < NPIX);
    const int py = p / WW, px = p % WW;

    u32 lda[2], ldb[2];
    lda[0] = lds_addr(&As[0][r * 64 + half * 32]);
    lda[1] = lds_addr(&As[1][r * 64 + half * 32]);
    ldb[0] = lds_addr(&Bs[0][r * 64 + half * 32]);
    ldb[1] = lds_addr(&Bs[1][r * 64 + half * 32]);

    v8f acc[2][4];
    #pragma unroll
    for (int mi = 0; mi < 2; ++mi)
        #pragma unroll
        for (int ni = 0; ni < 4; ++ni) acc[mi][ni] = (v8f)(0.0f);

    const int NSTEP = 9 * (CIN / 64);         // 144
    auto issue = [&](int step, int buf) {
        int tap = step >> 4;
        int cb  = (step & 15) * 64;
        int ty = tap / 3, tx = tap % 3;
        // invalid pixel rows read the all-zero padding pixel (0,0) of xpad
        const u16* asrc = pv ? (xpad + ((py + ty) * PW + (px + tx)) * CIN + cb + half * 32)
                             : (xpad + half * 32);
        const u16* bsrc = w1t + (tap * COUT + nbase + r) * CIN + cb + half * 32;
        async_copy64(lda[buf], asrc);
        async_copy64(ldb[buf], bsrc);
    };

    issue(0, 0);
    for (int step = 0; step < NSTEP; ++step) {
        const int cur = step & 1;
        wait_async0();             // my fills of buf[cur] have landed
        __syncthreads();           // everyone's fills landed; prev compute done
        if (step + 1 < NSTEP) issue(step + 1, cur ^ 1);  // overlap with WMMAs

        const u16* Ab = As[cur];
        const u16* Bb = Bs[cur];
        #pragma unroll
        for (int kk = 0; kk < 64; kk += 32) {
            v16bf af[2], bf[4];
            #pragma unroll
            for (int mi = 0; mi < 2; ++mi)
                af[mi] = load_a_frag(&Ab[(wm * 32 + mi * 16 + (lane & 15)) * 64 + kk], lane);
            #pragma unroll
            for (int ni = 0; ni < 4; ++ni)
                bf[ni] = load_b_frag(&Bb[(wn * 64 + ni * 16 + (lane & 15)) * 64 + kk], lane);
            #pragma unroll
            for (int mi = 0; mi < 2; ++mi)
                #pragma unroll
                for (int ni = 0; ni < 4; ++ni)
                    acc[mi][ni] = __builtin_amdgcn_wmma_f32_16x16x32_bf16(
                        false, af[mi], false, bf[ni], (short)0, acc[mi][ni], false, false);
        }
    }

    // epilogue: + bias, ReLU, store bf16 NHWC
    #pragma unroll
    for (int mi = 0; mi < 2; ++mi)
        #pragma unroll
        for (int ni = 0; ni < 4; ++ni) {
            int n = nbase + wn * 64 + ni * 16 + (lane & 15);
            float bv = bias[n];
            #pragma unroll
            for (int rr = 0; rr < 8; ++rr) {
                int m = pbase + wm * 32 + mi * 16 + (lane >> 4) * 8 + rr;
                if (m < NPIX) {
                    float v = acc[mi][ni][rr] + bv;
                    v = v > 0.0f ? v : 0.0f;
                    hout[m * COUT + n] = f2bf(v);
                }
            }
        }
}

// ---------------- GEMM2: fused 1x1 convs ----------------
// M=15200, N=96 (60 loc | 30 score | 6 pad), K=512. 8 waves x 16 rows.
__global__ __launch_bounds__(256)
void k_conv1x1_wmma(const u16* __restrict__ h, const u16* __restrict__ w2t,
                    const float* __restrict__ loc_b, const float* __restrict__ score_b,
                    const u16* __restrict__ zeros, float* __restrict__ out2) {
    __shared__ u16 As[2][128 * 64];
    __shared__ u16 Bs[2][N2 * 64];
    const int tid  = threadIdx.x;
    const int lane = tid & 31;
    const int wv   = tid >> 5;
    const int pbase = blockIdx.x * 128;
    const int r    = tid >> 1;
    const int half = tid & 1;
    const int p    = pbase + r;
    const bool pv  = (p < NPIX);
    const bool bstage = (tid < N2 * 2);

    u32 lda[2], ldb[2];
    lda[0] = lds_addr(&As[0][r * 64 + half * 32]);
    lda[1] = lds_addr(&As[1][r * 64 + half * 32]);
    ldb[0] = lds_addr(&Bs[0][(r < N2 ? r : 0) * 64 + half * 32]);
    ldb[1] = lds_addr(&Bs[1][(r < N2 ? r : 0) * 64 + half * 32]);

    v8f acc[6];
    #pragma unroll
    for (int ni = 0; ni < 6; ++ni) acc[ni] = (v8f)(0.0f);

    const int NSTEP = COUT / 64;               // 8
    auto issue = [&](int step, int buf) {
        int cb = step * 64;
        const u16* asrc = pv ? (h + p * COUT + cb + half * 32) : (zeros + half * 32);
        async_copy64(lda[buf], asrc);
        if (bstage)
            async_copy64(ldb[buf], w2t + r * COUT + cb + half * 32);
    };

    issue(0, 0);
    for (int step = 0; step < NSTEP; ++step) {
        const int cur = step & 1;
        wait_async0();
        __syncthreads();
        if (step + 1 < NSTEP) issue(step + 1, cur ^ 1);

        const u16* Ab = As[cur];
        const u16* Bb = Bs[cur];
        #pragma unroll
        for (int kk = 0; kk < 64; kk += 32) {
            v16bf af = load_a_frag(&Ab[(wv * 16 + (lane & 15)) * 64 + kk], lane);
            #pragma unroll
            for (int ni = 0; ni < 6; ++ni) {
                v16bf bfg = load_b_frag(&Bb[(ni * 16 + (lane & 15)) * 64 + kk], lane);
                acc[ni] = __builtin_amdgcn_wmma_f32_16x16x32_bf16(
                    false, af, false, bfg, (short)0, acc[ni], false, false);
            }
        }
    }

    #pragma unroll
    for (int ni = 0; ni < 6; ++ni) {
        int n = ni * 16 + (lane & 15);
        float bv = (n < 60) ? loc_b[n] : ((n < 90) ? score_b[n - 60] : 0.0f);
        #pragma unroll
        for (int rr = 0; rr < 8; ++rr) {
            int m = pbase + wv * 16 + (lane >> 4) * 8 + rr;
            if (m < NPIX) out2[m * N2 + n] = acc[ni][rr] + bv;
        }
    }
}

// ---------------- scatter rpn_locs / rpn_scores ----------------
__global__ void k_scatter(const float* __restrict__ out2, float* __restrict__ dout) {
    int t = blockIdx.x * 256 + threadIdx.x;
    if (t >= NPIX * 90) return;
    int p = t / 90, c = t % 90;
    float v = out2[p * N2 + c];
    if (c < 60) dout[OFF_LOCS + p * 60 + c] = v;
    else        dout[OFF_SCORES + p * 30 + (c - 60)] = v;
}

// ---------------- anchors + loc2bbox + clip + filter ----------------
__global__ void k_proposals(const float* __restrict__ out2,
                            const int* __restrict__ img_h, const int* __restrict__ img_w,
                            const int* __restrict__ scale,
                            float* __restrict__ dout, float* __restrict__ keys,
                            int* __restrict__ idxs, float* __restrict__ boxes) {
    int i = blockIdx.x * 256 + threadIdx.x;
    if (i >= SORTN) return;
    idxs[i] = i;
    if (i >= NANCH) { keys[i] = -__builtin_inff(); return; }
    int p = i / NA, a = i % NA;
    int y = p / WW, x = p % WW;
    const float ratios[3] = {0.5f, 1.0f, 2.0f};
    const float scales[5] = {2.0f, 4.0f, 8.0f, 16.0f, 32.0f};
    float rr = ratios[a / 5], ss = scales[a % 5];
    float hh = 16.0f * ss * sqrtf(rr);
    float ww = 16.0f * ss * sqrtf(1.0f / rr);
    float a0 = y * 16.0f + 8.0f - 0.5f * hh;
    float a1 = x * 16.0f + 8.0f - 0.5f * ww;
    float a2 = y * 16.0f + 8.0f + 0.5f * hh;
    float a3 = x * 16.0f + 8.0f + 0.5f * ww;
    dout[OFF_ANCHOR + i * 4 + 0] = a0;
    dout[OFF_ANCHOR + i * 4 + 1] = a1;
    dout[OFF_ANCHOR + i * 4 + 2] = a2;
    dout[OFF_ANCHOR + i * 4 + 3] = a3;
    float dy = out2[p * N2 + a * 4 + 0];
    float dx = out2[p * N2 + a * 4 + 1];
    float dh = out2[p * N2 + a * 4 + 2];
    float dw = out2[p * N2 + a * 4 + 3];
    float sh = a2 - a0, sw = a3 - a1;
    float cy = a0 + 0.5f * sh, cx = a1 + 0.5f * sw;
    float ncy = dy * sh + cy, ncx = dx * sw + cx;
    float nh = expf(dh) * sh, nw = expf(dw) * sw;
    float IH = (float)img_h[0], IW = (float)img_w[0];
    float r0 = fminf(fmaxf(ncy - 0.5f * nh, 0.0f), IH);
    float r1 = fminf(fmaxf(ncx - 0.5f * nw, 0.0f), IW);
    float r2 = fminf(fmaxf(ncy + 0.5f * nh, 0.0f), IH);
    float r3 = fminf(fmaxf(ncx + 0.5f * nw, 0.0f), IW);
    boxes[i * 4 + 0] = r0; boxes[i * 4 + 1] = r1;
    boxes[i * 4 + 2] = r2; boxes[i * 4 + 3] = r3;
    float msz = 16.0f * (float)scale[0];
    bool ok = ((r2 - r0) >= msz) && ((r3 - r1) >= msz);
    float fg = out2[p * N2 + 60 + a * 2 + 1];
    keys[i] = ok ? fg : -__builtin_inff();
}

// ---------------- bitonic sort step (descending) ----------------
__global__ void k_bitonic(float* __restrict__ keys, int* __restrict__ idxs, u32 k, u32 j) {
    u32 i = blockIdx.x * 256 + threadIdx.x;
    u32 l = i ^ j;
    if (l <= i || i >= SORTN) return;
    float ki = keys[i], kl = keys[l];
    bool up = ((i & k) == 0);
    bool sw = up ? (ki < kl) : (ki > kl);
    if (sw) {
        keys[i] = kl; keys[l] = ki;
        int t = idxs[i]; idxs[i] = idxs[l]; idxs[l] = t;
    }
}

// ---------------- gather top-6000 ----------------
__global__ void k_gather(const float* __restrict__ keys, const int* __restrict__ idxs,
                         const float* __restrict__ boxes,
                         float* __restrict__ tboxes, int* __restrict__ tvalid) {
    int t = blockIdx.x * 256 + threadIdx.x;
    if (t >= NPRE) return;
    int id = idxs[t];
    int cid = (id < NANCH) ? id : 0;
    tboxes[t * 4 + 0] = boxes[cid * 4 + 0];
    tboxes[t * 4 + 1] = boxes[cid * 4 + 1];
    tboxes[t * 4 + 2] = boxes[cid * 4 + 2];
    tboxes[t * 4 + 3] = boxes[cid * 4 + 3];
    tvalid[t] = (keys[t] != -__builtin_inff()) ? 1 : 0;
}

// ---------------- greedy NMS (single workgroup) ----------------
__global__ __launch_bounds__(256)
void k_nms(const float* __restrict__ tboxes, const int* __restrict__ tvalid,
           float* __restrict__ dout) {
    __shared__ unsigned char sup[NPRE];
    __shared__ int first;
    int tid = threadIdx.x;
    for (int i = tid; i < NPRE; i += 256) sup[i] = tvalid[i] ? 0 : 1;
    for (int i = tid; i < NPOST; i += 256) ((int*)dout)[OFF_RIDX + i] = 0;
    __syncthreads();
    for (int k = 0; k < NPOST; ++k) {
        if (tid == 0) first = 0x7fffffff;
        __syncthreads();
        int loc = 0x7fffffff;
        for (int i = tid; i < NPRE; i += 256)
            if (!sup[i]) { loc = i; break; }
        atomicMin(&first, loc);
        __syncthreads();
        int fi = first;
        if (fi < NPRE) {
            float b0 = tboxes[fi * 4 + 0], b1 = tboxes[fi * 4 + 1];
            float b2 = tboxes[fi * 4 + 2], b3 = tboxes[fi * 4 + 3];
            if (tid < 4) dout[OFF_ROIS + k * 4 + tid] = tboxes[fi * 4 + tid];
            float ai = (b2 - b0) * (b3 - b1);
            for (int i = tid; i < NPRE; i += 256) {
                float c0 = tboxes[i * 4 + 0], c1 = tboxes[i * 4 + 1];
                float c2 = tboxes[i * 4 + 2], c3 = tboxes[i * 4 + 3];
                float yy1 = fmaxf(b0, c0), xx1 = fmaxf(b1, c1);
                float yy2 = fminf(b2, c2), xx2 = fminf(b3, c3);
                float inter = fmaxf(yy2 - yy1, 0.0f) * fmaxf(xx2 - xx1, 0.0f);
                float area = (c2 - c0) * (c3 - c1);
                float iou = inter / (ai + area - inter + 1e-12f);
                if (iou >= 0.7f) sup[i] = 1;
            }
        } else {
            if (tid < 4) dout[OFF_ROIS + k * 4 + tid] = 0.0f;
        }
        __syncthreads();
    }
}

// ---------------- host launcher ----------------
extern "C" void kernel_launch(void* const* d_in, const int* in_sizes, int n_in,
                              void* d_out, int out_size, void* d_ws, size_t ws_size,
                              hipStream_t stream) {
    const float* x       = (const float*)d_in[0];
    const float* conv1_w = (const float*)d_in[1];
    const float* conv1_b = (const float*)d_in[2];
    const float* score_w = (const float*)d_in[3];
    const float* score_b = (const float*)d_in[4];
    const float* loc_w   = (const float*)d_in[5];
    const float* loc_b   = (const float*)d_in[6];
    const int*   img_h   = (const int*)d_in[7];
    const int*   img_w   = (const int*)d_in[8];
    const int*   scale   = (const int*)d_in[9];
    float* out = (float*)d_out;

    // workspace carve (256B aligned)
    char* w = (char*)d_ws;
    size_t off = 0;
    auto carve = [&](size_t bytes) {
        void* p = w + off;
        off = (off + bytes + 255) & ~(size_t)255;
        return p;
    };
    u16*   xpad   = (u16*)  carve((size_t)PH * PW * CIN * 2);
    u16*   w1t    = (u16*)  carve((size_t)9 * COUT * CIN * 2);
    u16*   hbuf   = (u16*)  carve((size_t)NPIX * COUT * 2);
    u16*   w2t    = (u16*)  carve((size_t)N2 * COUT * 2);
    float* out2   = (float*)carve((size_t)NPIX * N2 * 4);
    float* keys   = (float*)carve((size_t)SORTN * 4);
    int*   idxs   = (int*)  carve((size_t)SORTN * 4);
    float* boxes  = (float*)carve((size_t)NANCH * 4 * 4);
    float* tboxes = (float*)carve((size_t)NPRE * 4 * 4);
    int*   tvalid = (int*)  carve((size_t)NPRE * 4);
    (void)ws_size; (void)in_sizes; (void)n_in; (void)out_size;

    k_pad_x  <<<(PH * PW * CIN + 255) / 256, 256, 0, stream>>>(x, xpad);
    k_pack_w1<<<(9 * COUT * CIN + 255) / 256, 256, 0, stream>>>(conv1_w, w1t);
    k_pack_w2<<<(N2 * COUT + 255) / 256,      256, 0, stream>>>(loc_w, score_w, w2t);

    dim3 g1((NPIX + 127) / 128, COUT / 128);
    k_conv3x3_wmma<<<g1, 256, 0, stream>>>(xpad, w1t, conv1_b, hbuf);
    // xpad's first pixel (padding) is an all-zero 1024-ch row -> zero source
    k_conv1x1_wmma<<<(NPIX + 127) / 128, 256, 0, stream>>>(hbuf, w2t, loc_b, score_b, xpad, out2);

    k_scatter  <<<(NPIX * 90 + 255) / 256, 256, 0, stream>>>(out2, out);
    k_proposals<<<SORTN / 256, 256, 0, stream>>>(out2, img_h, img_w, scale, out, keys, idxs, boxes);

    for (u32 k = 2; k <= SORTN; k <<= 1)
        for (u32 j = k >> 1; j > 0; j >>= 1)
            k_bitonic<<<SORTN / 256, 256, 0, stream>>>(keys, idxs, k, j);

    k_gather<<<(NPRE + 255) / 256, 256, 0, stream>>>(keys, idxs, boxes, tboxes, tvalid);
    k_nms   <<<1, 256, 0, stream>>>(tboxes, tvalid, out);
}
